// xLSTM_33071248179634
// MI455X (gfx1250) — matrix operand order repaired and checked by reading, not compiled, not run on hardware
//
#include <hip/hip_runtime.h>

#define LL  2
#define BB  4
#define SS  512
#define DD  512
#define NHH 8
#define HDD 64
#define PP  1024
#define HH  512
#define KSS 4
#define MM  (SS*BB)   // 2048 rows

typedef __attribute__((ext_vector_type(16))) __bf16       bf16x16;
typedef __attribute__((ext_vector_type(8)))  float        floatx8;
typedef __attribute__((ext_vector_type(4)))  unsigned int u32x4;

union Frag { u32x4 u[2]; bf16x16 v; };

__device__ __forceinline__ unsigned short f2bf(float x){
  unsigned int u = __float_as_uint(x);
  u += 0x7fffu + ((u >> 16) & 1u);        // round-to-nearest-even
  return (unsigned short)(u >> 16);
}
__device__ __forceinline__ float sigmoidf_(float x){ return 1.0f/(1.0f + __expf(-x)); }
__device__ __forceinline__ float siluf_(float x){ return x * sigmoidf_(x); }

// ---------------------------------------------------------------- convert f32 -> bf16
__global__ __launch_bounds__(256)
void cvt_bf16_kernel(const float* __restrict__ in, unsigned short* __restrict__ out, int n){
  int i = blockIdx.x*256 + threadIdx.x;
  if (i < n) out[i] = f2bf(in[i]);
}

// ---------------------------------------------------------------- LayerNorm over D=512
__global__ __launch_bounds__(256)
void ln_kernel(const float* __restrict__ x, const float* __restrict__ g,
               const float* __restrict__ b, unsigned short* __restrict__ out_bf){
  const int row = blockIdx.x;
  const int tid = threadIdx.x;
  __shared__ float s1[256], s2[256];
  const float a0 = x[(size_t)row*DD + tid];
  const float a1 = x[(size_t)row*DD + 256 + tid];
  s1[tid] = a0 + a1; s2[tid] = a0*a0 + a1*a1;
  __syncthreads();
  for (int st = 128; st > 0; st >>= 1){
    if (tid < st){ s1[tid] += s1[tid+st]; s2[tid] += s2[tid+st]; }
    __syncthreads();
  }
  const float mu   = s1[0] * (1.0f/DD);
  const float var  = s2[0] * (1.0f/DD) - mu*mu;
  const float rstd = rsqrtf(var + 1e-5f);
  out_bf[(size_t)row*DD + tid]       = f2bf((a0 - mu)*rstd*g[tid]     + b[tid]);
  out_bf[(size_t)row*DD + 256 + tid] = f2bf((a1 - mu)*rstd*g[256+tid] + b[256+tid]);
}

// ---------------------------------------------------------------- WMMA GEMM (double-buffered)
// C[M x N] = A[M x K](bf16) @ Bw[N x K]^T(bf16) + bias[N] (+ residual[M x N])
// Tile: 128x128 per workgroup, BK=32, 8 waves each own a 16(M) x 128(N) strip.
#define BM   128
#define BN   128
#define BKK  32
#define LSTR 80                      // bytes/row in LDS: 32 bf16 = 64B + 16B pad
#define LBUF (2*BM*LSTR)             // one buffer = A tile + B tile = 20480 B

__global__ __launch_bounds__(256)
void gemm_bf16_wmma(const unsigned short* __restrict__ A, const unsigned short* __restrict__ Bw,
                    const float* __restrict__ bias, const float* __restrict__ residual,
                    float* __restrict__ Cout, unsigned short* __restrict__ Cbf,
                    int N, int K){
  __shared__ __align__(16) unsigned char lds[2*LBUF];   // 40960 B, double buffered
  const int bm   = blockIdx.x * BM;
  const int bn   = blockIdx.y * BN;
  const int tid  = threadIdx.x;
  const int wave = tid >> 5;
  const int lane = tid & 31;
  const int lr   = lane & 15;
  const int lg   = lane >> 4;

  // this thread's two 16B staging chunks (512 chunks cover a 128x32 bf16 tile)
  const int c0 = tid*2,      c1 = tid*2 + 1;
  const int r0 = c0 >> 2,    cc0 = c0 & 3;
  const int r1 = c1 >> 2,    cc1 = c1 & 3;
  const unsigned short* gA0 = A  + (size_t)(bm + r0)*K + cc0*8;
  const unsigned short* gA1 = A  + (size_t)(bm + r1)*K + cc1*8;
  const unsigned short* gB0 = Bw + (size_t)(bn + r0)*K + cc0*8;
  const unsigned short* gB1 = Bw + (size_t)(bn + r1)*K + cc1*8;

  floatx8 acc[8];
  floatx8 zero = {0.f,0.f,0.f,0.f,0.f,0.f,0.f,0.f};
  #pragma unroll
  for (int t = 0; t < 8; ++t) acc[t] = zero;

  // prologue: stage K-tile 0 into buffer 0
  u32x4 ga0 = *(const u32x4*)gA0;
  u32x4 ga1 = *(const u32x4*)gA1;
  u32x4 gb0 = *(const u32x4*)gB0;
  u32x4 gb1 = *(const u32x4*)gB1;
  *(u32x4*)(lds + r0*LSTR + cc0*16)            = ga0;
  *(u32x4*)(lds + r1*LSTR + cc1*16)            = ga1;
  *(u32x4*)(lds + BM*LSTR + r0*LSTR + cc0*16)  = gb0;
  *(u32x4*)(lds + BM*LSTR + r1*LSTR + cc1*16)  = gb1;
  __syncthreads();

  const int kTiles = K >> 5;
  const int arow = wave*16 + lr;
  for (int kt = 0; kt < kTiles; ++kt){
    unsigned char* cur = lds + (size_t)(kt & 1)*LBUF;
    unsigned char* nxt = lds + (size_t)((kt + 1) & 1)*LBUF;
    const bool more = (kt + 1) < kTiles;
    if (more){
      const size_t koff = (size_t)(kt + 1)*BKK;
      ga0 = *(const u32x4*)(gA0 + koff);
      ga1 = *(const u32x4*)(gA1 + koff);
      gb0 = *(const u32x4*)(gB0 + koff);
      gb1 = *(const u32x4*)(gB1 + koff);
    }
    // A fragment: 16x32 bf16 — lane group g: K = 8g..8g+7 then 16+8g..16+8g+7
    Frag af;
    af.u[0] = *(const u32x4*)(cur + arow*LSTR + lg*16);
    af.u[1] = *(const u32x4*)(cur + arow*LSTR + 32 + lg*16);
    // all 8 B fragments: 32x16 bf16 — lane group g: K = 8g..8g+7 then 16+8g..16+8g+7
    Frag bf_[8];
    #pragma unroll
    for (int t = 0; t < 8; ++t){
      const int brow = t*16 + lr;
      bf_[t].u[0] = *(const u32x4*)(cur + BM*LSTR + brow*LSTR + lg*16);
      bf_[t].u[1] = *(const u32x4*)(cur + BM*LSTR + brow*LSTR + 32 + lg*16);
    }
    #pragma unroll
    for (int t = 0; t < 8; ++t)
      acc[t] = __builtin_amdgcn_wmma_f32_16x16x32_bf16(
          false, af.v, false, bf_[t].v, (short)0, acc[t], false, false);
    if (more){
      *(u32x4*)(nxt + r0*LSTR + cc0*16)           = ga0;
      *(u32x4*)(nxt + r1*LSTR + cc1*16)           = ga1;
      *(u32x4*)(nxt + BM*LSTR + r0*LSTR + cc0*16) = gb0;
      *(u32x4*)(nxt + BM*LSTR + r1*LSTR + cc1*16) = gb1;
      __syncthreads();
    }
  }
  // epilogue: D layout — VGPR j: lanes 0-15 M=j, lanes 16-31 M=j+8; N=lane%16
  const int mbase = bm + wave*16 + ((lane >= 16) ? 8 : 0);
  #pragma unroll
  for (int t = 0; t < 8; ++t){
    const int col = bn + t*16 + lr;
    const float bv = bias[col];
    #pragma unroll
    for (int j = 0; j < 8; ++j){
      const int row = mbase + j;
      float val = acc[t][j] + bv;
      if (residual) val += residual[(size_t)row*N + col];
      Cout[(size_t)row*N + col] = val;
      if (Cbf) Cbf[(size_t)row*N + col] = f2bf(val);
    }
  }
}

// ---------------------------------------------------------------- causal conv over features + SiLU
__global__ __launch_bounds__(256)
void conv_silu_kernel(const float* __restrict__ xt, const float* __restrict__ cw,
                      const float* __restrict__ cbp,
                      float* __restrict__ xc, unsigned short* __restrict__ xc_bf){
  const int i   = blockIdx.x*256 + threadIdx.x;
  const int p   = i & (PP-1);
  float acc = cbp[0];
  #pragma unroll
  for (int j = 0; j < KSS; ++j){
    const int pp = p - (KSS-1) + j;
    if (pp >= 0) acc += cw[j] * xt[(size_t)i - p + pp];
  }
  const float s = siluf_(acc);
  xc[i]    = s;
  xc_bf[i] = f2bf(s);
}

// ---------------------------------------------------------------- i/f gate projections (N=8 each)
__global__ __launch_bounds__(256)
void gates_kernel(const float* __restrict__ xc, const float* __restrict__ Wi,
                  const float* __restrict__ bi, const float* __restrict__ Wf,
                  const float* __restrict__ bf_, float* __restrict__ ig, float* __restrict__ fg){
  const int row = blockIdx.x;
  const int tid = threadIdx.x;
  const int slot  = tid & 15;     // 0..7 -> i gate, 8..15 -> f gate
  const int chunk = tid >> 4;     // 16 chunks of 64 over K=1024
  __shared__ float part[256];
  const float* wrow = (slot < 8) ? (Wi + (size_t)slot*PP) : (Wf + (size_t)(slot-8)*PP);
  const float* xrow = xc + (size_t)row*PP;
  float p = 0.f;
  for (int kk = chunk*64; kk < chunk*64 + 64; ++kk) p += xrow[kk]*wrow[kk];
  part[tid] = p;
  __syncthreads();
  if (tid < 16){
    float s = 0.f;
    for (int c2 = 0; c2 < 16; ++c2) s += part[c2*16 + slot];
    if (slot < 8) ig[(size_t)row*NHH + slot]     = s + bi[slot];
    else          fg[(size_t)row*NHH + (slot-8)] = s + bf_[slot-8];
  }
}

// ---------------------------------------------------------------- sequential mLSTM state scan
// one workgroup per (b, head); C[64x64] in registers (16 floats/thread)
__global__ __launch_bounds__(256)
void scan_kernel(const float* __restrict__ q, const float* __restrict__ k,
                 const float* __restrict__ v, const float* __restrict__ ig,
                 const float* __restrict__ fg, const float* __restrict__ og,
                 float* __restrict__ h){
  const int b   = blockIdx.x >> 3;
  const int hh  = blockIdx.x & 7;
  const int tid = threadIdx.x;
  const int d   = tid >> 2;            // 0..63 row of C
  const int nn0 = (tid & 3) * 16;      // 16-wide column slice
  __shared__ float qv[64], kv[64], vv[64], nvec[64], part[256], denp[64], sc[4];
  float cr[16];
  #pragma unroll
  for (int j = 0; j < 16; ++j) cr[j] = 0.f;
  if (tid < 64) nvec[tid] = 1.f;       // n0 = ones
  if (tid == 0) sc[2] = 0.f;           // m0 = 0
  __syncthreads();
  for (int t = 0; t < SS; ++t){
    const size_t rowH = ((size_t)t*BB + b)*HH + hh*HDD;
    if (tid < 64){
      qv[tid] = q[rowH + tid];
      kv[tid] = k[rowH + tid] * 0.125f;   // 1/sqrt(HD)
      vv[tid] = v[rowH + tid];
    }
    if (tid == 0){
      const size_t rg = ((size_t)t*BB + b)*NHH + hh;
      const float iv = ig[rg], fv = fg[rg];
      const float m  = sc[2];
      const float mt = fmaxf(fv + m, iv);
      sc[0] = __expf(iv - mt);
      sc[1] = __expf(fv - mt + m);
      sc[2] = mt;
    }
    __syncthreads();
    const float ie = sc[0], fe = sc[1];
    const float vd = vv[d];
    float num = 0.f;
    #pragma unroll
    for (int j = 0; j < 16; ++j){
      cr[j] = fe*cr[j] + (ie*vd)*kv[nn0+j];
      num  += cr[j]*qv[nn0+j];
    }
    part[tid] = num;
    __syncthreads();
    if (tid < 64){
      const float nn_ = fe*nvec[tid] + ie*kv[tid];
      nvec[tid] = nn_;
      denp[tid] = nn_*qv[tid];
    }
    __syncthreads();
    if (tid == 0){
      float s = 0.f;
      for (int i2 = 0; i2 < 64; ++i2) s += denp[i2];
      sc[3] = fmaxf(fabsf(s), 1.0f);
    }
    __syncthreads();
    if (tid < 64){
      const float numd = part[tid*4] + part[tid*4+1] + part[tid*4+2] + part[tid*4+3];
      h[rowH + tid] = sigmoidf_(og[rowH + tid]) * numd / sc[3];
    }
    __syncthreads();
  }
}

// ---------------------------------------------------------------- GroupNorm + skip + SiLU(rt) mix
__global__ __launch_bounds__(256)
void post_kernel(const float* __restrict__ h, const float* __restrict__ gng,
                 const float* __restrict__ gnb, const float* __restrict__ skip,
                 const float* __restrict__ rt, unsigned short* __restrict__ mix_bf){
  const int row = blockIdx.x;
  const int tid = threadIdx.x;
  __shared__ float s1[8], s2[8];
  if (tid < 8){ s1[tid] = 0.f; s2[tid] = 0.f; }
  __syncthreads();
  const float x0 = h[(size_t)row*HH + tid];
  const float x1 = h[(size_t)row*HH + 256 + tid];
  const int g0 = tid >> 6;
  const int g1 = 4 + (tid >> 6);
  atomicAdd(&s1[g0], x0); atomicAdd(&s2[g0], x0*x0);
  atomicAdd(&s1[g1], x1); atomicAdd(&s2[g1], x1*x1);
  __syncthreads();
  {
    const float mu = s1[g0]*(1.f/HDD);
    const float rstd = rsqrtf(s2[g0]*(1.f/HDD) - mu*mu + 1e-5f);
    const int col = tid;
    const float xn  = (x0 - mu)*rstd*gng[col] + gnb[col];
    const float val = (xn + skip[(size_t)row*HH + col]) * siluf_(rt[(size_t)row*HH + col]);
    mix_bf[(size_t)row*HH + col] = f2bf(val);
  }
  {
    const float mu = s1[g1]*(1.f/HDD);
    const float rstd = rsqrtf(s2[g1]*(1.f/HDD) - mu*mu + 1e-5f);
    const int col = 256 + tid;
    const float xn  = (x1 - mu)*rstd*gng[col] + gnb[col];
    const float val = (xn + skip[(size_t)row*HH + col]) * siluf_(rt[(size_t)row*HH + col]);
    mix_bf[(size_t)row*HH + col] = f2bf(val);
  }
}

// ================================================================ launcher
extern "C" void kernel_launch(void* const* d_in, const int* in_sizes, int n_in,
                              void* d_out, int out_size, void* d_ws, size_t ws_size,
                              hipStream_t stream){
  const float* x      = (const float*)d_in[0];
  const float* ln_g   = (const float*)d_in[1];
  const float* ln_b   = (const float*)d_in[2];
  const float* Wl     = (const float*)d_in[3];
  const float* bl     = (const float*)d_in[4];
  const float* Wr     = (const float*)d_in[5];
  const float* br     = (const float*)d_in[6];
  const float* conv_w = (const float*)d_in[7];
  const float* conv_b = (const float*)d_in[8];
  const float* Wq     = (const float*)d_in[9];
  const float* bq     = (const float*)d_in[10];
  const float* Wk     = (const float*)d_in[11];
  const float* bk     = (const float*)d_in[12];
  const float* Wv     = (const float*)d_in[13];
  const float* bv     = (const float*)d_in[14];
  const float* Wi     = (const float*)d_in[15];
  const float* bi     = (const float*)d_in[16];
  const float* Wf     = (const float*)d_in[17];
  const float* bfp    = (const float*)d_in[18];
  const float* Wo     = (const float*)d_in[19];
  const float* bo     = (const float*)d_in[20];
  const float* gn_g   = (const float*)d_in[21];
  const float* gn_b   = (const float*)d_in[22];
  const float* skw    = (const float*)d_in[23];
  const float* skb    = (const float*)d_in[24];
  const float* dww    = (const float*)d_in[25];
  const float* dwb    = (const float*)d_in[26];
  (void)in_sizes; (void)n_in; (void)out_size; (void)ws_size;

  char* ws = (char*)d_ws;
  size_t off = 0;
  auto alloc = [&](size_t bytes)->char*{
    char* p = ws + off; off += (bytes + 255) & ~(size_t)255; return p;
  };
  float* inp             = (float*)alloc((size_t)MM*DD*4);
  unsigned short* xn_bf  = (unsigned short*)alloc((size_t)MM*DD*2);
  float* xt              = (float*)alloc((size_t)MM*PP*4);
  unsigned short* xt_bf  = (unsigned short*)alloc((size_t)MM*PP*2);
  float* xc              = (float*)alloc((size_t)MM*PP*4);
  unsigned short* xc_bf  = (unsigned short*)alloc((size_t)MM*PP*2);
  float* rt              = (float*)alloc((size_t)MM*HH*4);
  float* qb              = (float*)alloc((size_t)MM*HH*4);
  float* kb              = (float*)alloc((size_t)MM*HH*4);
  float* vb              = (float*)alloc((size_t)MM*HH*4);
  float* ogb             = (float*)alloc((size_t)MM*HH*4);
  float* skipb           = (float*)alloc((size_t)MM*HH*4);
  float* igb             = (float*)alloc((size_t)MM*NHH*4);
  float* fgb             = (float*)alloc((size_t)MM*NHH*4);
  float* hb              = (float*)alloc((size_t)MM*HH*4);
  unsigned short* mix_bf = (unsigned short*)alloc((size_t)MM*HH*2);
  unsigned short* wbuf   = (unsigned short*)alloc((size_t)PP*DD*2); // reused per weight

  for (int l = 0; l < LL; ++l){
    const float* lin = (l == 0) ? x : inp;
    float* lout      = (l == LL-1) ? (float*)d_out : inp;

    ln_kernel<<<MM, 256, 0, stream>>>(lin, ln_g + l*DD, ln_b + l*DD, xn_bf);

    cvt_bf16_kernel<<<(PP*DD+255)/256, 256, 0, stream>>>(Wl + (size_t)l*PP*DD, wbuf, PP*DD);
    gemm_bf16_wmma<<<dim3(MM/128, PP/128), 256, 0, stream>>>(
        xn_bf, wbuf, bl + l*PP, nullptr, xt, xt_bf, PP, DD);

    cvt_bf16_kernel<<<(HH*DD+255)/256, 256, 0, stream>>>(Wr + (size_t)l*HH*DD, wbuf, HH*DD);
    gemm_bf16_wmma<<<dim3(MM/128, HH/128), 256, 0, stream>>>(
        xn_bf, wbuf, br + l*HH, nullptr, rt, nullptr, HH, DD);

    conv_silu_kernel<<<(MM*PP)/256, 256, 0, stream>>>(xt, conv_w + l*KSS, conv_b + l, xc, xc_bf);

    cvt_bf16_kernel<<<(HH*PP+255)/256, 256, 0, stream>>>(Wq + (size_t)l*HH*PP, wbuf, HH*PP);
    gemm_bf16_wmma<<<dim3(MM/128, HH/128), 256, 0, stream>>>(
        xc_bf, wbuf, bq + l*HH, nullptr, qb, nullptr, HH, PP);

    cvt_bf16_kernel<<<(HH*PP+255)/256, 256, 0, stream>>>(Wk + (size_t)l*HH*PP, wbuf, HH*PP);
    gemm_bf16_wmma<<<dim3(MM/128, HH/128), 256, 0, stream>>>(
        xc_bf, wbuf, bk + l*HH, nullptr, kb, nullptr, HH, PP);

    cvt_bf16_kernel<<<(HH*PP+255)/256, 256, 0, stream>>>(Wv + (size_t)l*HH*PP, wbuf, HH*PP);
    gemm_bf16_wmma<<<dim3(MM/128, HH/128), 256, 0, stream>>>(
        xt_bf, wbuf, bv + l*HH, nullptr, vb, nullptr, HH, PP);

    cvt_bf16_kernel<<<(HH*PP+255)/256, 256, 0, stream>>>(Wo + (size_t)l*HH*PP, wbuf, HH*PP);
    gemm_bf16_wmma<<<dim3(MM/128, HH/128), 256, 0, stream>>>(
        xt_bf, wbuf, bo + l*HH, nullptr, ogb, nullptr, HH, PP);

    cvt_bf16_kernel<<<(HH*PP+255)/256, 256, 0, stream>>>(skw + (size_t)l*HH*PP, wbuf, HH*PP);
    gemm_bf16_wmma<<<dim3(MM/128, HH/128), 256, 0, stream>>>(
        xc_bf, wbuf, skb + l*HH, nullptr, skipb, nullptr, HH, PP);

    gates_kernel<<<MM, 256, 0, stream>>>(xc, Wi + (size_t)l*NHH*PP, bi + l*NHH,
                                         Wf + (size_t)l*NHH*PP, bfp + l*NHH, igb, fgb);

    scan_kernel<<<BB*NHH, 256, 0, stream>>>(qb, kb, vb, igb, fgb, ogb, hb);

    post_kernel<<<MM, 256, 0, stream>>>(hb, gn_g + l*HH, gn_b + l*HH, skipb, rt, mix_bf);

    cvt_bf16_kernel<<<(DD*HH+255)/256, 256, 0, stream>>>(dww + (size_t)l*DD*HH, wbuf, DD*HH);
    gemm_bf16_wmma<<<dim3(MM/128, DD/128), 256, 0, stream>>>(
        mix_bf, wbuf, dwb + l*DD, lin, lout, nullptr, DD, HH);
  }
}
